// Segment_3D_8057358647894
// MI455X (gfx1250) — matrix operand order repaired and checked by reading, not compiled, not run on hardware
//
#include <hip/hip_runtime.h>
#include <hip/hip_bf16.h>

typedef __attribute__((ext_vector_type(16))) _Float16 v16h;
typedef __attribute__((ext_vector_type(8)))  _Float16 v8h;
typedef __attribute__((ext_vector_type(8)))  float    v8f;

#define WMMA_F16(a, b, c) __builtin_amdgcn_wmma_f32_16x16x32_f16(false, (a), false, (b), (short)0, (c), false, false)

// ---------------------------------------------------------------------------
// Prepack weights into per-lane WMMA A-fragment layout (16x32 f16 per step).
// K ordering: K = tap*I + i  (input channel fastest).
// A layout: lanes 0-15: j=0..15 -> K=32s+j
//           lanes16-31: j=0..15 -> K=32s+16+j
// src layout: w[o][i][tap] flat = (o*I + i)*ntaps + tap ; zero-pad tap >= ntaps.
// ---------------------------------------------------------------------------
__global__ void prepack_a(const float* __restrict__ src, int I, int ntaps,
                          int nsteps, _Float16* __restrict__ dst) {
    int idx = blockIdx.x * 256 + threadIdx.x;
    int total = nsteps * 512;
    if (idx >= total) return;
    int step = idx >> 9;
    int r    = idx & 511;
    int lane = r >> 4;
    int j    = r & 15;
    int m    = lane & 15;
    int K;
    if (lane < 16) K = 32 * step + j;
    else           K = 32 * step + 16 + j;
    int tap = K / I;
    int i   = K % I;
    float v = (tap < ntaps) ? src[(m * I + i) * ntaps + tap] : 0.0f;
    dst[idx] = (_Float16)v;
}

// Packed bias+ReLU+f16 store of one 16x16 D fragment half (8 channels/lane).
__device__ __forceinline__ void store_d(const v8f& acc, const float* __restrict__ bias,
                                        int mb, _Float16* __restrict__ dst) {
    v8h o8;
#pragma unroll
    for (int v = 0; v < 8; ++v) {
        float t = acc[v] + bias[mb + v];
        o8[v] = (_Float16)(t > 0.0f ? t : 0.0f);
    }
    *(v8h*)(dst + mb) = o8;
}

// ---------------------------------------------------------------------------
// conv1: 3x3x3, 1 -> 16 channels, input x f32 (B,128,32,32), output SA f16
// NDHWC [b][d][h][w][c16]. One wave = one full 32-wide row (2 accumulators).
// ---------------------------------------------------------------------------
__global__ void __launch_bounds__(256)
conv1_wmma(const float* __restrict__ x, const _Float16* __restrict__ apack,
           const float* __restrict__ bias, _Float16* __restrict__ out) {
    const int bid = blockIdx.x;
    const int hb = bid & 3, d = (bid >> 2) & 127, b = bid >> 9;
    __shared__ alignas(32) float xt[3 * 10 * 34];   // [dz][hy][wx]
    const int tid = threadIdx.x;
    for (int pix = tid; pix < 3 * 10 * 34; pix += 256) {
        int wx = pix % 34, hy = (pix / 34) % 10, dz = pix / 340;
        int din = d - 1 + dz, hin = hb * 8 - 1 + hy, win = wx - 1;
        float v = 0.0f;
        if (din >= 0 && din < 128 && hin >= 0 && hin < 32 && win >= 0 && win < 32)
            v = x[(((size_t)b * 128 + din) * 32 + hin) * 32 + win];
        xt[pix] = v;
    }
    __syncthreads();
    const int lane = tid & 31, h_local = tid >> 5;
    const int n = lane & 15, hi16 = lane >> 4;
    v16h a = *(const v16h*)(apack + lane * 16);
    v16h bf0 = {}, bf1 = {};
    const int kb = hi16 * 16;
#pragma unroll
    for (int j = 0; j < 16; ++j) {
        int tap = kb + j;
        if (tap < 27) {
            int kd = tap / 9, kh = (tap / 3) % 3, kw = tap % 3;
            int base = (kd * 10 + h_local + kh) * 34 + n + kw;
            bf0[j] = (_Float16)xt[base];
            bf1[j] = (_Float16)xt[base + 16];
        }
    }
    v8f acc0 = {}, acc1 = {};
    acc0 = WMMA_F16(a, bf0, acc0);
    acc1 = WMMA_F16(a, bf1, acc1);
    const int mb = hi16 * 8;
    size_t p = (((size_t)b * 128 + d) * 32 + hb * 8 + h_local) * 32 + n;
    store_d(acc0, bias, mb, out + p * 16);
    store_d(acc1, bias, mb, out + (p + 16) * 16);
}

// ---------------------------------------------------------------------------
// conv3d: 3x3x3, 16 -> 16 channels, f16 NDHWC in/out. K = 27*16 padded to 448
// (14 steps x 2 independent WMMA accumulators per wave). Haloed input tile
// [3][10][34][16] f16 staged in LDS via global_load_async_to_lds_b128.
// ---------------------------------------------------------------------------
__global__ void __launch_bounds__(256)
conv3d_wmma(const _Float16* __restrict__ in, const _Float16* __restrict__ apack,
            const float* __restrict__ bias, _Float16* __restrict__ out) {
    const int bid = blockIdx.x;
    const int hb = bid & 3, d = (bid >> 2) & 127, b = bid >> 9;
    __shared__ alignas(32) _Float16 st[3 * 10 * 34 * 16];  // 32640 B
    const int tid = threadIdx.x;
    for (int pix = tid; pix < 3 * 10 * 34; pix += 256) {
        int wx = pix % 34, hy = (pix / 34) % 10, dz = pix / 340;
        int din = d - 1 + dz, hin = hb * 8 - 1 + hy, win = wx - 1;
        if (din >= 0 && din < 128 && hin >= 0 && hin < 32 && win >= 0 && win < 32) {
            unsigned lds = (unsigned)(uintptr_t)(st + pix * 16);
            unsigned long long g = (unsigned long long)(uintptr_t)
                (in + (((size_t)b * 128 + din) * 1024 + hin * 32 + win) * 16);
            // 32 bytes per pixel: two async b128, offset applies to LDS and global
            asm volatile("global_load_async_to_lds_b128 %0, %1, off\n\t"
                         "global_load_async_to_lds_b128 %0, %1, off offset:16"
                         :: "v"(lds), "v"(g) : "memory");
        } else {
            uint4 z; z.x = z.y = z.z = z.w = 0u;
            uint4* dst = (uint4*)(st + pix * 16);
            dst[0] = z; dst[1] = z;
        }
    }
    asm volatile("s_wait_asynccnt 0x0" ::: "memory");
    __syncthreads();
    const int lane = tid & 31, h_local = tid >> 5;
    const int n = lane & 15, hi16 = lane >> 4;
    v8f acc0 = {}, acc1 = {};
#pragma unroll
    for (int step = 0; step < 14; ++step) {
        v16h a = *(const v16h*)(apack + (step * 32 + lane) * 16);
        int tap = step * 2 + hi16;
        v16h bf0 = {}, bf1 = {};
        if (tap < 27) {
            int kd = tap / 9, kh = (tap / 3) % 3, kw = tap % 3;
            const _Float16* p = st + ((kd * 10 + h_local + kh) * 34 + n + kw) * 16;
            bf0 = *(const v16h*)(p);
            bf1 = *(const v16h*)(p + 256);   // +16 pixels in w
        }
        acc0 = WMMA_F16(a, bf0, acc0);
        acc1 = WMMA_F16(a, bf1, acc1);
    }
    const int mb = hi16 * 8;
    size_t p = (((size_t)b * 128 + d) * 32 + hb * 8 + h_local) * 32 + n;
    store_d(acc0, bias, mb, out + p * 16);
    store_d(acc1, bias, mb, out + (p + 16) * 16);
}

// ---------------------------------------------------------------------------
// cpr: out2d[b,o,h,w] = relu(sum_{i,d} SA[b,i,d,h,w] * wc[o,i,d] + bc[o]).
// K = 2048 (d,i) i-fastest == SA's NDHWC layout: B fragments load straight
// from global. One wave = one 32-wide row, 64 steps x 2 accumulators.
// Output: f16 NHWC [b][h][w][c16].
// ---------------------------------------------------------------------------
__global__ void __launch_bounds__(256)
cpr_wmma(const _Float16* __restrict__ in, const _Float16* __restrict__ apack,
         const float* __restrict__ bias, _Float16* __restrict__ out) {
    const int tid = threadIdx.x;
    const int lane = tid & 31, wid = tid >> 5;
    const int gid = blockIdx.x * 8 + wid;     // 0..511
    const int b = gid >> 5, h = gid & 31;
    const int n = lane & 15, hi16 = lane >> 4;
    v8f acc0 = {}, acc1 = {};
    for (int step = 0; step < 64; ++step) {
        v16h a = *(const v16h*)(apack + (step * 32 + lane) * 16);
        int dcol = step * 2 + hi16;
        const _Float16* p = in + (((size_t)b * 128 + dcol) * 1024 + h * 32 + n) * 16;
        v16h bf0 = *(const v16h*)(p);
        v16h bf1 = *(const v16h*)(p + 256);
        acc0 = WMMA_F16(a, bf0, acc0);
        acc1 = WMMA_F16(a, bf1, acc1);
    }
    const int mb = hi16 * 8;
    size_t p = (((size_t)b * 32 + h) * 32 + n);
    store_d(acc0, bias, mb, out + p * 16);
    store_d(acc1, bias, mb, out + (p + 16) * 16);
}

// ---------------------------------------------------------------------------
// conv2d: 3x3, 16 -> 16 channels, f16 NHWC in/out. K = 144 padded to 160
// (5 steps x 2 accumulators). Haloed tile [10][34][16] f16 via async-to-LDS.
// ---------------------------------------------------------------------------
__global__ void __launch_bounds__(256)
conv2d_wmma(const _Float16* __restrict__ in, const _Float16* __restrict__ apack,
            const float* __restrict__ bias, _Float16* __restrict__ out) {
    const int bid = blockIdx.x;
    const int hb = bid & 3, b = bid >> 2;
    __shared__ alignas(32) _Float16 st[10 * 34 * 16];  // 10880 B
    const int tid = threadIdx.x;
    for (int pix = tid; pix < 10 * 34; pix += 256) {
        int wx = pix % 34, hy = pix / 34;
        int hin = hb * 8 - 1 + hy, win = wx - 1;
        if (hin >= 0 && hin < 32 && win >= 0 && win < 32) {
            unsigned lds = (unsigned)(uintptr_t)(st + pix * 16);
            unsigned long long g = (unsigned long long)(uintptr_t)
                (in + (((size_t)b * 32 + hin) * 32 + win) * 16);
            asm volatile("global_load_async_to_lds_b128 %0, %1, off\n\t"
                         "global_load_async_to_lds_b128 %0, %1, off offset:16"
                         :: "v"(lds), "v"(g) : "memory");
        } else {
            uint4 z; z.x = z.y = z.z = z.w = 0u;
            uint4* dst = (uint4*)(st + pix * 16);
            dst[0] = z; dst[1] = z;
        }
    }
    asm volatile("s_wait_asynccnt 0x0" ::: "memory");
    __syncthreads();
    const int lane = tid & 31, h_local = tid >> 5;
    const int n = lane & 15, hi16 = lane >> 4;
    v8f acc0 = {}, acc1 = {};
#pragma unroll
    for (int step = 0; step < 5; ++step) {
        v16h a = *(const v16h*)(apack + (step * 32 + lane) * 16);
        int tap = step * 2 + hi16;
        v16h bf0 = {}, bf1 = {};
        if (tap < 9) {
            int kh = tap / 3, kw = tap % 3;
            const _Float16* p = st + ((h_local + kh) * 34 + n + kw) * 16;
            bf0 = *(const v16h*)(p);
            bf1 = *(const v16h*)(p + 256);
        }
        acc0 = WMMA_F16(a, bf0, acc0);
        acc1 = WMMA_F16(a, bf1, acc1);
    }
    const int mb = hi16 * 8;
    size_t p = (((size_t)b * 32 + hb * 8 + h_local) * 32 + n);
    store_d(acc0, bias, mb, out + p * 16);
    store_d(acc1, bias, mb, out + (p + 16) * 16);
}

// ---------------------------------------------------------------------------
// stage: SEG==1 so argmax/mask path is dead -> global mean pool + 16x16 GEMM.
// feat: f16 NHWC (b,32,32,16). m[b][o] = bo[o] + sum_c mean_c * wo[o][c].
// ---------------------------------------------------------------------------
__global__ void __launch_bounds__(256)
stage_kernel(const _Float16* __restrict__ feat, const float* __restrict__ wo,
             const float* __restrict__ bo, float* __restrict__ m) {
    const int b = blockIdx.x, tid = threadIdx.x;
    __shared__ float part[16][16];
    __shared__ float pool[16];
    const int c = tid & 15, seg = tid >> 4;
    const _Float16* fb = feat + (size_t)b * 1024 * 16;
    float s = 0.0f;
    for (int p = seg * 64; p < seg * 64 + 64; ++p) s += (float)fb[p * 16 + c];
    part[seg][c] = s;
    __syncthreads();
    if (tid < 16) {
        float t = 0.0f;
#pragma unroll
        for (int k = 0; k < 16; ++k) t += part[k][tid];
        pool[tid] = t * (1.0f / 1024.0f);
    }
    __syncthreads();
    if (tid < 16) {
        float t = bo[tid];
#pragma unroll
        for (int k = 0; k < 16; ++k) t += pool[k] * wo[tid * 16 + k];
        m[b * 16 + tid] = t;
    }
}

__global__ void fuse_kernel(const float* __restrict__ fp,
                            const float* __restrict__ m1, const float* __restrict__ m2,
                            const float* __restrict__ m3, const float* __restrict__ m4,
                            const float* __restrict__ m5, float* __restrict__ out) {
    const int i = threadIdx.x;  // 256 = 16*16
    float a1 = m1[i], a2 = m2[i], a3 = m3[i], a4 = m4[i], a5 = m5[i];
    out[i] = fp[0] * a1 + fp[1] * a2 + fp[2] * a3 + fp[3] * a4 + fp[4] * a5;
    out[256 + i] = a1;
    out[512 + i] = a2;
    out[768 + i] = a3;
    out[1024 + i] = a4;
    out[1280 + i] = a5;
}

// ---------------------------------------------------------------------------
extern "C" void kernel_launch(void* const* d_in, const int* in_sizes, int n_in,
                              void* d_out, int out_size, void* d_ws, size_t ws_size,
                              hipStream_t stream) {
    (void)in_sizes; (void)n_in; (void)out_size; (void)ws_size;
    const float* x   = (const float*)d_in[0];
    const float* w1  = (const float*)d_in[1];  const float* b1  = (const float*)d_in[2];
    const float* w2  = (const float*)d_in[3];  const float* b2  = (const float*)d_in[4];
    const float* w3  = (const float*)d_in[5];  const float* b3  = (const float*)d_in[6];
    const float* wc1 = (const float*)d_in[7];  const float* bc1 = (const float*)d_in[8];
    const float* wc2 = (const float*)d_in[9];  const float* bc2 = (const float*)d_in[10];
    const float* wc3 = (const float*)d_in[11]; const float* bc3 = (const float*)d_in[12];
    const float* w4  = (const float*)d_in[13]; const float* b4  = (const float*)d_in[14];
    const float* w5  = (const float*)d_in[15]; const float* b5  = (const float*)d_in[16];
    // d_in[17..26] = ws*/bs* : dead (SEG==1 makes the mask all-ones)
    const float* wo1 = (const float*)d_in[27]; const float* bo1 = (const float*)d_in[28];
    const float* wo2 = (const float*)d_in[29]; const float* bo2 = (const float*)d_in[30];
    const float* wo3 = (const float*)d_in[31]; const float* bo3 = (const float*)d_in[32];
    const float* wo4 = (const float*)d_in[33]; const float* bo4 = (const float*)d_in[34];
    const float* wo5 = (const float*)d_in[35]; const float* bo5 = (const float*)d_in[36];
    const float* fp  = (const float*)d_in[37];

    // Workspace carve-up (halfs). SA buffers: 16*128*32*32*16 = 33,554,432 halfs.
    const size_t SA_H = (size_t)16 * 128 * 32 * 32 * 16;
    const size_t F_H  = (size_t)16 * 32 * 32 * 16;  // 262144
    _Float16* SAa = (_Float16*)d_ws;
    _Float16* SAb = SAa + SA_H;
    _Float16* F1  = SAb + SA_H;
    _Float16* F2  = F1 + F_H;
    _Float16* F3  = F2 + F_H;
    _Float16* F4  = F3 + F_H;
    _Float16* F5  = F4 + F_H;
    _Float16* ap1  = F5 + F_H;         // 1  step  * 512
    _Float16* ap2  = ap1 + 512;        // 14 steps * 512 = 7168
    _Float16* ap3  = ap2 + 7168;
    _Float16* apc1 = ap3 + 7168;       // 64 steps * 512 = 32768
    _Float16* apc2 = apc1 + 32768;
    _Float16* apc3 = apc2 + 32768;
    _Float16* ap4  = apc3 + 32768;     // 5 steps * 512 = 2560
    _Float16* ap5  = ap4 + 2560;
    float* m1 = (float*)(ap5 + 2560);
    float* m2 = m1 + 256; float* m3 = m2 + 256; float* m4 = m3 + 256; float* m5 = m4 + 256;

    // Weight pre-swizzle into A-fragment layout
    prepack_a<<<2,   256, 0, stream>>>(w1,  1,  27, 1,  ap1);
    prepack_a<<<28,  256, 0, stream>>>(w2,  16, 27, 14, ap2);
    prepack_a<<<28,  256, 0, stream>>>(w3,  16, 27, 14, ap3);
    prepack_a<<<128, 256, 0, stream>>>(wc1, 16, 128, 64, apc1);
    prepack_a<<<128, 256, 0, stream>>>(wc2, 16, 128, 64, apc2);
    prepack_a<<<128, 256, 0, stream>>>(wc3, 16, 128, 64, apc3);
    prepack_a<<<10,  256, 0, stream>>>(w4,  16, 9,  5,  ap4);
    prepack_a<<<10,  256, 0, stream>>>(w5,  16, 9,  5,  ap5);

    // Pipeline (SA1/SA2 ping-pong stays L2-resident: 2 x 67 MB < 192 MB L2)
    conv1_wmma <<<8192, 256, 0, stream>>>(x,   ap1,  b1,  SAa);   // SA1
    cpr_wmma   <<<64,   256, 0, stream>>>(SAa, apc1, bc1, F1);    // SA1_2D
    conv3d_wmma<<<8192, 256, 0, stream>>>(SAa, ap2,  b2,  SAb);   // SA2
    cpr_wmma   <<<64,   256, 0, stream>>>(SAb, apc2, bc2, F2);    // SA2_2D
    conv3d_wmma<<<8192, 256, 0, stream>>>(SAb, ap3,  b3,  SAa);   // SA3 (reuse SAa)
    cpr_wmma   <<<64,   256, 0, stream>>>(SAa, apc3, bc3, F3);    // SA3_2D
    conv2d_wmma<<<64,   256, 0, stream>>>(F3,  ap4,  b4,  F4);    // SA4_2D
    conv2d_wmma<<<64,   256, 0, stream>>>(F4,  ap5,  b5,  F5);    // SA5_2D

    stage_kernel<<<16, 256, 0, stream>>>(F1, wo1, bo1, m1);
    stage_kernel<<<16, 256, 0, stream>>>(F2, wo2, bo2, m2);
    stage_kernel<<<16, 256, 0, stream>>>(F3, wo3, bo3, m3);
    stage_kernel<<<16, 256, 0, stream>>>(F4, wo4, bo4, m4);
    stage_kernel<<<16, 256, 0, stream>>>(F5, wo5, bo5, m5);

    fuse_kernel<<<1, 256, 0, stream>>>(fp, m1, m2, m3, m4, m5, (float*)d_out);
}
